// SchNet_64819646431978
// MI455X (gfx1250) — matrix operand order repaired and checked by reading, not compiled, not run on hardware
//
#include <hip/hip_runtime.h>
#include <hip/hip_bf16.h>
#include <math.h>

// SchNet forward for MI455X (gfx1250): bf16 WMMA GEMMs w/ f32 accumulation,
// fused edge kernel (RBF -> filter MLP -> gather/modulate/scatter-add).
// Round 3: templated node GEMM (compile-time C/ACT/BIAS/RES -> immediate-offset
// epilogue, no 64-bit per-element muls), vectorized aux kernels.

typedef __attribute__((ext_vector_type(16))) __bf16 v16bf;
typedef __attribute__((ext_vector_type(8)))  float  v8f;

struct U8x32 { uint4 lo, hi; };   // 32-byte carrier for v16bf bit_cast

#define N_ATOMS 50000
#define N_EDGES 800000
#define HCH 128
#define RBF 64
#define NBLK 2
#define CUTF 5.0f

// Branch-free tanh: 1 - 2/(exp(2x)+1). exp->inf => 1, exp->0 => -1.
__device__ __forceinline__ float fast_tanh(float x) {
  return 1.0f - 2.0f * __builtin_amdgcn_rcpf(__expf(2.0f * x) + 1.0f);
}

// Pack 4 f32 -> 4 bf16, single 8-byte LDS store.
__device__ __forceinline__ void stage4(__bf16* dst, float4 f) {
  union { __bf16 b[4]; uint2 u; } p;
  p.b[0] = (__bf16)f.x; p.b[1] = (__bf16)f.y;
  p.b[2] = (__bf16)f.z; p.b[3] = (__bf16)f.w;
  *reinterpret_cast<uint2*>(dst) = p.u;
}

// ---- WMMA fragment loaders (layouts per CDNA5 ISA 7.12.2, wave32) ----

// A: 16x32 bf16 from row-major LDS [16][K]. Per lane the fragment is two
// contiguous 16B chunks: k = kbase+half*8 .. +7 and kbase+16+half*8 .. +7.
__device__ __forceinline__ v16bf load_a_frag(const __bf16* sA, int K, int kbase, int lane) {
  int m = lane & 15, half8 = (lane >> 4) << 3;
  const __bf16* p = sA + m * K + kbase + half8;
  U8x32 u;
  u.lo = *reinterpret_cast<const uint4*>(p);        // ds_load_b128
  u.hi = *reinterpret_cast<const uint4*>(p + 16);   // ds_load_b128
  return __builtin_bit_cast(v16bf, u);
}

// B: 32x16 bf16 from row-major LDS [K][C], columns col0..col0+15.
// lane half h: elems 0-7 K=kbase+8h..+7, elems 8-15 K=kbase+16+8h..+7. Pair {2v,2v+1}.
// Only executed once per kernel (hoisted), so strided u16 reads are fine.
__device__ __forceinline__ v16bf load_b_frag(const __bf16* sW, int C, int kbase, int col, int lane) {
  int k0 = kbase + ((lane >> 4) << 3);
  v16bf b;
#pragma unroll
  for (int v = 0; v < 8; ++v) {
    int kk = k0 + 2 * v + (v >= 4 ? 8 : 0);
    b[2 * v]     = sW[kk * C + col];
    b[2 * v + 1] = sW[(kk + 1) * C + col];
  }
  return b;
}

// ---- Node GEMM: out[N,C] = op(A[N,128] @ W[128,C] + bias) (+resid), K=128 ----
template <int C, bool ACT, bool HAS_BIAS, bool HAS_RES>
__global__ __launch_bounds__(256) void node_gemm_kernel(
    const float* __restrict__ A, const float* __restrict__ W,
    const float* __restrict__ bias, const float* __restrict__ resid,
    float* __restrict__ out, int nrows) {
  __shared__ __bf16 sW[128 * C];
  __shared__ __bf16 sA[16 * 128];
  int tid = threadIdx.x;
#pragma unroll
  for (int i = 0; i < (128 * C) / 1024; ++i)
    stage4(sW + i * 1024 + tid * 4, *(const float4*)(W + i * 1024 + tid * 4));
  int wave = tid >> 5, lane = tid & 31;
  int n = lane & 15, mofs = (lane < 16) ? 0 : 8;
  int col0 = wave * 16, col = col0 + n;
  float bv = (HAS_BIAS && col0 < C) ? bias[col] : 0.f;
  __syncthreads();
  v16bf bfr[4];                       // loop-invariant B fragments (32 VGPRs)
  if (col0 < C) {
#pragma unroll
    for (int s = 0; s < 4; ++s) bfr[s] = load_b_frag(sW, C, 32 * s, col, lane);
  }
  int ntiles = nrows >> 4;
  for (int tile = blockIdx.x; tile < ntiles; tile += gridDim.x) {
    int row0 = tile << 4;
#pragma unroll
    for (int i = 0; i < 2; ++i)
      stage4(sA + i * 1024 + tid * 4, *(const float4*)(A + (size_t)row0 * 128 + i * 1024 + tid * 4));
    __syncthreads();
    if (col0 < C) {
      v8f c = {};
#pragma unroll
      for (int s = 0; s < 4; ++s)
        c = __builtin_amdgcn_wmma_f32_16x16x32_bf16(false, load_a_frag(sA, 128, 32 * s, lane),
                                                    false, bfr[s], (short)0, c, false, false);
      size_t base = (size_t)(row0 + mofs) * C + col;
      float* po = out + base;
      const float* pr = resid + base;
#pragma unroll
      for (int r = 0; r < 8; ++r) {          // immediate offsets r*C*4
        float v = c[r] + bv;
        if (ACT) v = fast_tanh(v);
        if (HAS_RES) v += pr[(size_t)r * C];
        po[(size_t)r * C] = v;
      }
    }
    __syncthreads();
  }
}

// ---- Fused edge kernel: RBF -> (tanh(ea@W1+b1))@W2+b2 -> h[src]*W -> atomic agg[dst] ----
__global__ __launch_bounds__(256) void edge_kernel(
    const float* __restrict__ dist, const int* __restrict__ eidx,
    const float* __restrict__ fW1, const float* __restrict__ fb1,
    const float* __restrict__ fW2, const float* __restrict__ fb2,
    const float* __restrict__ h, float* __restrict__ agg, int nE) {
  __shared__ __bf16 sW1[RBF * HCH];   // 16 KB
  __shared__ __bf16 sW2[HCH * HCH];   // 32 KB
  __shared__ __bf16 sA[16 * RBF];     // edge_attr tile (2 KB)
  __shared__ __bf16 sT[16 * HCH];     // tanh intermediate (4 KB)
  __shared__ int    sSrc[16], sDst[16];
  __shared__ float  sDist[16], sEnv[16];
  int tid = threadIdx.x;
#pragma unroll
  for (int i = 0; i < (RBF * HCH) / 1024; ++i)
    stage4(sW1 + i * 1024 + tid * 4, *(const float4*)(fW1 + i * 1024 + tid * 4));
#pragma unroll
  for (int i = 0; i < (HCH * HCH) / 1024; ++i)
    stage4(sW2 + i * 1024 + tid * 4, *(const float4*)(fW2 + i * 1024 + tid * 4));
  int lane = tid & 31;
  int n = lane & 15, mofs = (lane < 16) ? 0 : 8;
  int col = (tid >> 5) * 16 + n;
  float b1v = fb1[col], b2v = fb2[col];
  const float step  = CUTF / (RBF - 1);
  const float coeff = -0.5f / (step * step);
  __syncthreads();
  v16bf bfr1[2], bfr2[4];             // loop-invariant B fragments (48 VGPRs)
#pragma unroll
  for (int s = 0; s < 2; ++s) bfr1[s] = load_b_frag(sW1, HCH, 32 * s, col, lane);
#pragma unroll
  for (int s = 0; s < 4; ++s) bfr2[s] = load_b_frag(sW2, HCH, 32 * s, col, lane);
  int ntiles = nE >> 4;
  for (int tile = blockIdx.x; tile < ntiles; tile += gridDim.x) {
    int e0 = tile << 4;
    if (tid < 16) {
      int e = e0 + tid;
      float d = dist[e];
      sDist[tid] = d;
      sEnv[tid]  = 0.5f * (__cosf(3.14159265358979f * d / CUTF) + 1.0f) * (d < CUTF ? 1.f : 0.f);
      sSrc[tid]  = eidx[e];
      sDst[tid]  = eidx[nE + e];
    }
    if (tid == 32 && tile + gridDim.x < ntiles)
      __builtin_prefetch(&dist[(tile + gridDim.x) << 4], 0, 0);   // global_prefetch_b8
    __syncthreads();
    // Gaussian RBF * cosine cutoff: 16x64 tile, 4 consecutive elems/thread
    {
      int idx = tid * 4;
      int e = idx >> 6, r = idx & 63;
      float d0 = sDist[e], ev = sEnv[e];
      float4 f;
      float t0 = d0 - step * (float)(r + 0); f.x = ev * __expf(coeff * t0 * t0);
      float t1 = d0 - step * (float)(r + 1); f.y = ev * __expf(coeff * t1 * t1);
      float t2 = d0 - step * (float)(r + 2); f.z = ev * __expf(coeff * t2 * t2);
      float t3 = d0 - step * (float)(r + 3); f.w = ev * __expf(coeff * t3 * t3);
      stage4(sA + idx, f);
    }
    __syncthreads();
    // GEMM1: [16x64]@[64x128], +b1, tanh -> sT
    {
      v8f c = {};
#pragma unroll
      for (int s = 0; s < 2; ++s)
        c = __builtin_amdgcn_wmma_f32_16x16x32_bf16(false, load_a_frag(sA, RBF, 32 * s, lane),
                                                    false, bfr1[s], (short)0, c, false, false);
#pragma unroll
      for (int r = 0; r < 8; ++r)
        sT[(mofs + r) * HCH + col] = (__bf16)fast_tanh(c[r] + b1v);
    }
    __syncthreads();
    // GEMM2: [16x128]@[128x128], +b2; gather h[src], modulate, scatter-add
    {
      v8f c = {};
#pragma unroll
      for (int s = 0; s < 4; ++s)
        c = __builtin_amdgcn_wmma_f32_16x16x32_bf16(false, load_a_frag(sT, HCH, 32 * s, lane),
                                                    false, bfr2[s], (short)0, c, false, false);
#pragma unroll
      for (int r = 0; r < 8; ++r) {
        int el = mofs + r;
        float msg = h[((size_t)sSrc[el] << 7) + col] * (c[r] + b2v);
        atomicAdd(&agg[((size_t)sDst[el] << 7) + col], msg);
      }
    }
    __syncthreads();
  }
}

__global__ void embed_kernel(const int* __restrict__ types, const float* __restrict__ table,
                             float* __restrict__ x, int n) {
  int i = blockIdx.x * blockDim.x + threadIdx.x;   // over n*32 float4s
  if (i < n * (HCH / 4)) {
    int a = i >> 5, c4 = i & 31;
    ((float4*)x)[i] = ((const float4*)(table + (size_t)types[a] * HCH))[c4];
  }
}

__global__ void zero_kernel(float4* __restrict__ p, int n4) {
  int i = blockIdx.x * blockDim.x + threadIdx.x;
  if (i < n4) p[i] = float4{0.f, 0.f, 0.f, 0.f};
}

__global__ void out_dot_kernel(const float* __restrict__ o1, const float* __restrict__ w2,
                               const float* __restrict__ b2, float* __restrict__ out, int n) {
  int i = blockIdx.x * blockDim.x + threadIdx.x;
  if (i < n) {
    const float4* p = (const float4*)(o1 + (size_t)i * 64);
    const float4* w = (const float4*)w2;
    float s = b2[0];
#pragma unroll
    for (int c = 0; c < 16; ++c) {
      float4 a = p[c], b = w[c];
      s += a.x * b.x + a.y * b.y + a.z * b.z + a.w * b.w;
    }
    out[i] = s;
  }
}

extern "C" void kernel_launch(void* const* d_in, const int* in_sizes, int n_in,
                              void* d_out, int out_size, void* d_ws, size_t ws_size,
                              hipStream_t stream) {
  (void)in_sizes; (void)n_in; (void)out_size; (void)ws_size;
  const int*   types = (const int*)d_in[0];
  const int*   eidx  = (const int*)d_in[1];
  const float* dist  = (const float*)d_in[2];
  const float* table = (const float*)d_in[3];
  const float* fW1   = (const float*)d_in[4];
  const float* fb1   = (const float*)d_in[5];
  const float* fW2   = (const float*)d_in[6];
  const float* fb2   = (const float*)d_in[7];
  const float* cW1   = (const float*)d_in[8];
  const float* cW2   = (const float*)d_in[9];
  const float* cb2   = (const float*)d_in[10];
  const float* bW    = (const float*)d_in[11];
  const float* bb    = (const float*)d_in[12];
  const float* oW1   = (const float*)d_in[13];
  const float* ob1   = (const float*)d_in[14];
  const float* oW2   = (const float*)d_in[15];
  const float* ob2   = (const float*)d_in[16];
  float* out = (float*)d_out;

  // Workspace: x[N,128] | h[N,128] | agg[N,128]  (agg reused as [N,64] for output MLP)
  float* x    = (float*)d_ws;
  float* hbuf = x    + (size_t)N_ATOMS * HCH;
  float* agg  = hbuf + (size_t)N_ATOMS * HCH;

  embed_kernel<<<(N_ATOMS * 32 + 255) / 256, 256, 0, stream>>>(types, table, x, N_ATOMS);
  for (int l = 0; l < NBLK; ++l) {
    // h = x @ conv_W1
    node_gemm_kernel<HCH, false, false, false><<<1024, 256, 0, stream>>>(
        x, cW1 + (size_t)l * HCH * HCH, nullptr, nullptr, hbuf, N_ATOMS);
    zero_kernel<<<(N_ATOMS * 32 + 255) / 256, 256, 0, stream>>>((float4*)agg, N_ATOMS * 32);
    // agg[dst] += h[src] * Filter(dist)
    edge_kernel<<<2048, 256, 0, stream>>>(dist, eidx,
                                          fW1 + (size_t)l * RBF * HCH, fb1 + (size_t)l * HCH,
                                          fW2 + (size_t)l * HCH * HCH, fb2 + (size_t)l * HCH,
                                          hbuf, agg, N_EDGES);
    // u = tanh(agg @ conv_W2 + conv_b2)   (reuse hbuf)
    node_gemm_kernel<HCH, true, true, false><<<1024, 256, 0, stream>>>(
        agg, cW2 + (size_t)l * HCH * HCH, cb2 + (size_t)l * HCH, nullptr, hbuf, N_ATOMS);
    // x = x + u @ blk_W + blk_b
    node_gemm_kernel<HCH, false, true, true><<<1024, 256, 0, stream>>>(
        hbuf, bW + (size_t)l * HCH * HCH, bb + (size_t)l * HCH, x, x, N_ATOMS);
  }
  // o1 = tanh(x @ out_W1 + out_b1)   [N,64] (reuse agg)
  node_gemm_kernel<64, true, true, false><<<1024, 256, 0, stream>>>(
      x, oW1, ob1, nullptr, agg, N_ATOMS);
  // energy = o1 @ out_W2 + out_b2
  out_dot_kernel<<<(N_ATOMS + 255) / 256, 256, 0, stream>>>(agg, oW2, ob2, out, N_ATOMS);
}
